// ODERNN_78537771975175
// MI455X (gfx1250) — 8-bit matrix operand order repaired and checked by reading, compiled, not run on hardware
//
#include <hip/hip_runtime.h>
#include <hip/hip_bf16.h>

typedef __attribute__((ext_vector_type(8)))  int   v8i;
typedef __attribute__((ext_vector_type(16))) int   v16i;
typedef __attribute__((ext_vector_type(8)))  float v8f;

#define BDIM 256
#define KDIM 512
#define UDIM 16
#define PDIM 64
#define HDIM 256
#define LDIM 64
#define THDIM 128
#define G3H  768

#define OFF_WHH   0
#define OFF_WIH   196608
#define OFF_WHEAD 245760
#define OFF_WLIFT 278528
#define OFF_H8    286720
#define OFF_X8    290816
#define OFF_FEAT  292864
#define OFF_THETA 294912
#define OFF_JUMP  303104
#define OFF_BIH   307200
#define OFF_BHH   310272
#define OFF_BHEAD 313344
#define OFF_BLIFT 313856
#define OFF_U     314112
#define OFF_DT    315136
#define LDS_BYTES 315200

#define WMMA64(A,B,C)  __builtin_amdgcn_wmma_f32_16x16x64_fp8_fp8((A),(B),(short)0,(C),false,false)
#define WMMA128(A,B,C) __builtin_amdgcn_wmma_f32_16x16x128_fp8_fp8((A),(B),(short)0,(C),false,false)

__device__ __forceinline__ float fast_rcp(float x) {
#if __has_builtin(__builtin_amdgcn_rcpf)
  return __builtin_amdgcn_rcpf(x);
#else
  return 1.0f / x;
#endif
}

__device__ __forceinline__ float fast_tanh(float x) {
#if __has_builtin(__builtin_amdgcn_tanhf)
  return __builtin_amdgcn_tanhf(x);
#else
  float e = __expf(2.0f * x);
  return 1.0f - 2.0f * fast_rcp(e + 1.0f);
#endif
}

__device__ __forceinline__ float sigf(float x) {
#if __has_builtin(__builtin_amdgcn_tanhf)
  return __builtin_amdgcn_tanhf(x * 0.5f) * 0.5f + 0.5f;
#else
  return fast_rcp(1.0f + __expf(-x));
#endif
}

__device__ __forceinline__ unsigned char f32_to_fp8(float f) {
  unsigned int u = __float_as_uint(f);
  unsigned int s = (u >> 24) & 0x80u;
  unsigned int a = u & 0x7FFFFFFFu;
  if (a < 0x3C800000u) {
    float af = __uint_as_float(a);
    unsigned int dc = (unsigned int)(af * 512.0f + 0.5f);
    return (unsigned char)(s | dc);
  }
  unsigned int r = a + 0x7FFFFu + ((a >> 20) & 1u);
  int v = (int)(r >> 20) - 960;
  if (v > 0x7E) v = 0x7E;
  return (unsigned char)(s | (unsigned int)v);
}

__device__ __forceinline__ void store_fp8_pair(char* p0, char* p1, float a, float b) {
#if __has_builtin(__builtin_amdgcn_cvt_pk_fp8_f32)
  int pk = __builtin_amdgcn_cvt_pk_fp8_f32(a, b, 0, false);
  *p0 = (char)(pk & 0xFF);
  *p1 = (char)((pk >> 8) & 0xFF);
#else
  *p0 = (char)f32_to_fp8(a);
  *p1 = (char)f32_to_fp8(b);
#endif
}

__device__ __forceinline__ v8i lds_a64(const char* base, int stride, int k0, int lane) {
  const char* row = base + (lane & 15) * stride + k0 + ((lane >> 4) << 4);
  uint2 q0 = *(const uint2*)(row +  0);
  uint2 q1 = *(const uint2*)(row +  8);
  uint2 q2 = *(const uint2*)(row + 32);
  uint2 q3 = *(const uint2*)(row + 40);
  v8i a;
  a[0]=(int)q0.x; a[1]=(int)q0.y; a[2]=(int)q1.x; a[3]=(int)q1.y;
  a[4]=(int)q2.x; a[5]=(int)q2.y; a[6]=(int)q3.x; a[7]=(int)q3.y;
  return a;
}
__device__ __forceinline__ v16i lds_a128(const char* base, int stride, int k0, int lane) {
  v8i lo = lds_a64(base, stride, k0,      lane);
  v8i hi = lds_a64(base, stride, k0 + 64, lane);
  v16i a;
#pragma unroll
  for (int i = 0; i < 8; ++i) { a[i] = lo[i]; a[8 + i] = hi[i]; }
  return a;
}
__device__ __forceinline__ v8i lds_b64(const char* wbase, int stride, int n0, int k0, int lane) {
  const char* row = wbase + (n0 + (lane & 15)) * stride + k0 + ((lane >> 4) << 4);
  int4 q0 = *(const int4*)(row +  0);
  int4 q1 = *(const int4*)(row + 32);
  v8i b;
  b[0]=q0.x; b[1]=q0.y; b[2]=q0.z; b[3]=q0.w;
  b[4]=q1.x; b[5]=q1.y; b[6]=q1.z; b[7]=q1.w;
  return b;
}
__device__ __forceinline__ v16i lds_b128(const char* wbase, int stride, int n0, int k0, int lane) {
  const char* row = wbase + (n0 + (lane & 15)) * stride + k0 + ((lane >> 4) << 4);
  int4 q0 = *(const int4*)(row +  0);
  int4 q1 = *(const int4*)(row + 32);
  int4 q2 = *(const int4*)(row + 64);
  int4 q3 = *(const int4*)(row + 96);
  v16i b;
  b[0]=q0.x;  b[1]=q0.y;  b[2]=q0.z;  b[3]=q0.w;
  b[4]=q1.x;  b[5]=q1.y;  b[6]=q1.z;  b[7]=q1.w;
  b[8]=q2.x;  b[9]=q2.y;  b[10]=q2.z; b[11]=q2.w;
  b[12]=q3.x; b[13]=q3.y; b[14]=q3.z; b[15]=q3.w;
  return b;
}

__global__ void __launch_bounds__(256)
odernn_fp8_kernel(const float* __restrict__ y0,     const float* __restrict__ u_seq,
                  const float* __restrict__ dt_seq, const float* __restrict__ W_lift,
                  const float* __restrict__ b_lift, const float* __restrict__ W_ih,
                  const float* __restrict__ b_ih,   const float* __restrict__ W_hh,
                  const float* __restrict__ b_hh,   const float* __restrict__ W_head,
                  const float* __restrict__ b_head, const float* __restrict__ jump,
                  float* __restrict__ y_out,        float* __restrict__ th_out)
{
  extern __shared__ char smem[];
  const int t     = threadIdx.x;
  const int lane  = t & 31;
  const int wid   = t >> 5;
  const int bbase = blockIdx.x * 16;

  char* Whh8   = smem + OFF_WHH;
  char* Wih8   = smem + OFF_WIH;
  char* Whead8 = smem + OFF_WHEAD;
  char* Wlift8 = smem + OFF_WLIFT;
  char* H8     = smem + OFF_H8;
  char* X8     = smem + OFF_X8;
  char* F8     = smem + OFF_FEAT;
  float* thetaL = (float*)(smem + OFF_THETA);
  float* jumpL  = (float*)(smem + OFF_JUMP);
  float* bihL   = (float*)(smem + OFF_BIH);
  float* bhhL   = (float*)(smem + OFF_BHH);
  float* bheadL = (float*)(smem + OFF_BHEAD);
  float* bliftL = (float*)(smem + OFF_BLIFT);
  float* uL     = (float*)(smem + OFF_U);
  float* dtL    = (float*)(smem + OFF_DT);

  for (int i = t; i < G3H * HDIM;  i += 256) Whh8[i]   = f32_to_fp8(W_hh[i]);
  for (int i = t; i < G3H * LDIM;  i += 256) Wih8[i]   = f32_to_fp8(W_ih[i]);
  for (int i = t; i < THDIM * HDIM; i += 256) Whead8[i] = f32_to_fp8(W_head[i]);
  for (int i = t; i < LDIM * 128;  i += 256) Wlift8[i] = 0;
  for (int i = t; i < 16 * 256;    i += 256) H8[i] = 0;
  for (int i = t; i < 16 * 128;    i += 256) { X8[i] = 0; F8[i] = 0; }
  for (int i = t; i < UDIM * PDIM; i += 256) jumpL[i] = jump[i];
  for (int i = t; i < G3H;         i += 256) { bihL[i] = b_ih[i]; bhhL[i] = b_hh[i]; }
  if (t < THDIM) bheadL[t] = b_head[t];
  if (t < LDIM)  bliftL[t] = b_lift[t];
  __syncthreads();
  for (int i = t; i < LDIM * 80; i += 256) {
    int n = i / 80, kk = i - n * 80;
    Wlift8[n * 128 + kk] = f32_to_fp8(W_lift[i]);
  }
  __syncthreads();

  const int pcol  = t & 63;
  const int mrow0 = (t >> 6) * 4;
  float yv[4];
#pragma unroll
  for (int r = 0; r < 4; ++r)
    yv[r] = y0[(size_t)(bbase + mrow0 + r) * PDIM + pcol];

  v8f hold[2];
#pragma unroll
  for (int i = 0; i < 8; ++i) { hold[0][i] = 0.0f; hold[1][i] = 0.0f; }

  const int cmb = (lane >> 4) * 8;
  const int cn  = lane & 15;

  for (int k = 0; k < KDIM; ++k) {
    {
      int m = t >> 4, uu = t & 15;
      float uval = u_seq[((size_t)(bbase + m) * KDIM + k) * UDIM + uu];
      uL[m * 16 + uu] = uval;
      F8[m * 128 + uu] = f32_to_fp8(uval);
      if (t < 16) dtL[t] = dt_seq[(size_t)(bbase + t) * KDIM + k];
#pragma unroll
      for (int r = 0; r < 4; r += 2)
        store_fp8_pair(&F8[(mrow0 + r)     * 128 + 16 + pcol],
                       &F8[(mrow0 + r + 1) * 128 + 16 + pcol],
                       yv[r], yv[r + 1]);
    }
    __syncthreads();

    if (wid < 4) {
      int n0 = wid * 16;
      v16i fa = lds_a128(F8, 128, 0, lane);
      v16i wb = lds_b128(Wlift8, 128, n0, 0, lane);
      v8f c = {};
      c = WMMA128(fa, wb, c);
      float bn = bliftL[n0 + cn];
      float xs[8];
#pragma unroll
      for (int i = 0; i < 8; ++i) {
        float v = c[i] + bn;
        xs[i] = v * sigf(v);
      }
#pragma unroll
      for (int i = 0; i < 8; i += 2)
        store_fp8_pair(&X8[(cmb + i)     * 128 + n0 + cn],
                       &X8[(cmb + i + 1) * 128 + n0 + cn],
                       xs[i], xs[i + 1]);
    }
    __syncthreads();

    v8i  xa  = lds_a64 (X8, 128, 0,   lane);
    v16i ha0 = lds_a128(H8, 256, 0,   lane);
    v16i ha1 = lds_a128(H8, 256, 128, lane);
    __syncthreads();

#pragma unroll
    for (int tt = 0; tt < 2; ++tt) {
      int j0 = wid * 32 + tt * 16;
      v8f gir = {}, giz = {}, gin = {}, ghr = {}, ghz = {}, ghn = {};
      gir = WMMA64 (xa,  lds_b64 (Wih8,  64,       j0, 0,   lane), gir);
      ghr = WMMA128(ha0, lds_b128(Whh8, 256,       j0, 0,   lane), ghr);
      ghr = WMMA128(ha1, lds_b128(Whh8, 256,       j0, 128, lane), ghr);
      giz = WMMA64 (xa,  lds_b64 (Wih8,  64, 256 + j0, 0,   lane), giz);
      ghz = WMMA128(ha0, lds_b128(Whh8, 256, 256 + j0, 0,   lane), ghz);
      ghz = WMMA128(ha1, lds_b128(Whh8, 256, 256 + j0, 128, lane), ghz);
      gin = WMMA64 (xa,  lds_b64 (Wih8,  64, 512 + j0, 0,   lane), gin);
      ghn = WMMA128(ha0, lds_b128(Whh8, 256, 512 + j0, 0,   lane), ghn);
      ghn = WMMA128(ha1, lds_b128(Whh8, 256, 512 + j0, 128, lane), ghn);

      int j = j0 + cn;
      float bir = bihL[j],       bhr = bhhL[j];
      float biz = bihL[256 + j], bhz = bhhL[256 + j];
      float bin = bihL[512 + j], bhn = bhhL[512 + j];
      float hnv[8];
#pragma unroll
      for (int i = 0; i < 8; ++i) {
        float rr = sigf(gir[i] + ghr[i] + bir + bhr);
        float zz = sigf(giz[i] + ghz[i] + biz + bhz);
        float nn = fast_tanh(gin[i] + bin + rr * (ghn[i] + bhn));
        float hn = (1.0f - zz) * nn + zz * hold[tt][i];
        hold[tt][i] = hn;
        hnv[i] = hn;
      }
#pragma unroll
      for (int i = 0; i < 8; i += 2)
        store_fp8_pair(&H8[(cmb + i)     * 256 + j],
                       &H8[(cmb + i + 1) * 256 + j],
                       hnv[i], hnv[i + 1]);
    }
    __syncthreads();

    {
      int n0 = wid * 16;
      v16i hb0 = lds_a128(H8, 256, 0,   lane);
      v16i hb1 = lds_a128(H8, 256, 128, lane);
      v8f c = {};
      c = WMMA128(hb0, lds_b128(Whead8, 256, n0, 0,   lane), c);
      c = WMMA128(hb1, lds_b128(Whead8, 256, n0, 128, lane), c);
      float bn = bheadL[n0 + cn];
#pragma unroll
      for (int i = 0; i < 8; ++i) {
        float th = 0.001f + 1.999f * sigf(c[i] + bn);
        thetaL[(cmb + i) * 128 + n0 + cn] = th;
      }
    }
    __syncthreads();

#pragma unroll
    for (int r = 0; r < 4; ++r) {
      int m = mrow0 + r;
      float acc = yv[r];
#pragma unroll
      for (int uu = 0; uu < 16; ++uu)
        acc += uL[m * 16 + uu] * jumpL[uu * 64 + pcol];
      float aa = thetaL[m * 128 + pcol];
      float dd = thetaL[m * 128 + 64 + pcol];
      float h  = dtL[m];
      float k1 = dd - aa * acc;
      float k2 = dd - aa * (acc + 0.5f * h * k1);
      float k3 = dd - aa * (acc + 0.5f * h * k2);
      float k4 = dd - aa * (acc + h * k3);
      float yn = acc + h * (1.0f / 6.0f) * (k1 + 2.0f * k2 + 2.0f * k3 + k4);
      yn = fmaxf(yn, 0.0f);
      yv[r] = yn;
      y_out[((size_t)(bbase + m) * KDIM + k) * PDIM + pcol] = yn;
    }
#pragma unroll
    for (int i = 0; i < 8; ++i) {
      int idx = t + i * 256;
      int m = idx >> 7, n = idx & 127;
      th_out[((size_t)(bbase + m) * KDIM + k) * THDIM + n] = thetaL[idx];
    }
    __syncthreads();
  }
}

extern "C" void kernel_launch(void* const* d_in, const int* in_sizes, int n_in,
                              void* d_out, int out_size, void* d_ws, size_t ws_size,
                              hipStream_t stream) {
  const float* y0     = (const float*)d_in[0];
  const float* u_seq  = (const float*)d_in[1];
  const float* dt_seq = (const float*)d_in[2];
  const float* W_lift = (const float*)d_in[3];
  const float* b_lift = (const float*)d_in[4];
  const float* W_ih   = (const float*)d_in[5];
  const float* b_ih   = (const float*)d_in[6];
  const float* W_hh   = (const float*)d_in[7];
  const float* b_hh   = (const float*)d_in[8];
  const float* W_head = (const float*)d_in[9];
  const float* b_head = (const float*)d_in[10];
  const float* jump   = (const float*)d_in[11];

  float* y_out  = (float*)d_out;
  float* th_out = y_out + (size_t)BDIM * KDIM * PDIM;

  (void)hipFuncSetAttribute((const void*)odernn_fp8_kernel,
                            hipFuncAttributeMaxDynamicSharedMemorySize, LDS_BYTES);

  odernn_fp8_kernel<<<BDIM / 16, 256, LDS_BYTES, stream>>>(
      y0, u_seq, dt_seq, W_lift, b_lift, W_ih, b_ih, W_hh, b_hh,
      W_head, b_head, jump, y_out, th_out);
}
